// RffGradPen_29145648071173
// MI455X (gfx1250) — matrix operand order repaired and checked by reading, not compiled, not run on hardware
//
#include <hip/hip_runtime.h>

typedef float    v2f  __attribute__((ext_vector_type(2)));
typedef float    v8f  __attribute__((ext_vector_type(8)));
typedef _Float16 v16h __attribute__((ext_vector_type(16)));
typedef _Float16 v8h  __attribute__((ext_vector_type(8)));

#define N_OBS   2048
#define DIM_IN  64
#define KH      1024  // hidden units

// -sqrt(2/1024)
#define NEG_SCALE (-0.044194173824159216f)

// ---------------------------------------------------------------------------
// Kernel 0: wT[d][l] = w[l][d]   (64 x 1024, 256 KB) for coalesced expansion
// ---------------------------------------------------------------------------
__global__ __launch_bounds__(256) void k_wt(const float* __restrict__ w,
                                            float* __restrict__ wT) {
    int i = blockIdx.x * 256 + threadIdx.x;   // 0 .. 65535
    int d = i >> 10;
    int l = i & 1023;
    wT[i] = w[l * DIM_IN + d];
}

// ---------------------------------------------------------------------------
// Kernel 1: sT[k][n] = -scale * sin( (x @ w^T)[n][k] + b[k] )   stored as f16
// One wave per 16x16 tile of z, f32 WMMA over K=64 (16 steps of K=4).
// ---------------------------------------------------------------------------
__global__ __launch_bounds__(256) void k_s(const float* __restrict__ x,
                                           const float* __restrict__ w,
                                           const float* __restrict__ b,
                                           _Float16* __restrict__ sT) {
    const int wave = threadIdx.x >> 5;
    const int lane = threadIdx.x & 31;
    const int tile = blockIdx.x * 8 + wave;   // 8192 tiles = (2048/16)*(1024/16)
    const int tn   = tile >> 6;               // n-tile 0..127
    const int tk   = tile & 63;               // k-tile 0..63
    const int n0   = tn * 16, k0 = tk * 16;
    const int row  = lane & 15;               // M (A) / N (B,C,D) index
    const int half = lane >> 4;

    // A tile: A[m][kk] = x[n0+m][d0+kk]; 32-bit A 16x4: lane<16 -> K=c, else K=c+2
    const float* xr = x + (size_t)(n0 + row) * DIM_IN + half * 2;
    // B tile: B[kk][nn] = w[k0+nn][d0+kk]; lane holds N=lane%16, K=2*half+c
    const float* wr = w + (size_t)(k0 + row) * DIM_IN + half * 2;

    v8f c = {};
    #pragma unroll
    for (int d0 = 0; d0 < DIM_IN; d0 += 4) {
        v2f a;  a.x = xr[d0];  a.y = xr[d0 + 1];
        v2f bb; bb.x = wr[d0]; bb.y = wr[d0 + 1];
        c = __builtin_amdgcn_wmma_f32_16x16x4_f32(false, a, false, bb,
                                                  (short)0, c, false, false);
    }

    const float bias = b[k0 + row];           // C col N = lane%16 -> hidden unit
    v8h sv;
    #pragma unroll
    for (int r = 0; r < 8; ++r) {             // C row M = r + 8*half -> obs n
        float z = c[r] + bias;
        sv[r] = (_Float16)(NEG_SCALE * __sinf(z));
    }
    // transposed store: sT[k0+row][n0 + 8*half + r], contiguous 16B per lane
    *(v8h*)(sT + (size_t)(k0 + row) * N_OBS + n0 + half * 8) = sv;
}

// ---------------------------------------------------------------------------
// Kernel 2: M[k][l] = (1/n) * sum_n sT[k][n]*sT[l][n]
// One wave per 32x32 tile (2x2 register blocking): 4 accumulators, 2 A-tiles,
// 2 B-tiles per K-step -> 4 WMMA per load group, 16 FLOP/byte from L2.
// f16 WMMA 16x16x32, f32 accumulate, 64 K-steps.
// ---------------------------------------------------------------------------
__global__ __launch_bounds__(256) void k_gram(const _Float16* __restrict__ sT,
                                              float* __restrict__ M) {
    const int wave = threadIdx.x >> 5;
    const int lane = threadIdx.x & 31;
    const int tile = blockIdx.x * 8 + wave;   // 1024 tiles = 32*32
    const int tk   = tile >> 5;
    const int tl   = tile & 31;
    const int k0   = tk * 32, l0 = tl * 32;
    const int row  = lane & 15;
    const int half = lane >> 4;

    // 16-bit A 16x32: lane holds M=lane%16; halves h: K = 16*half + h
    //   -> two contiguous 8-half runs at n0 + 16*half and n0 + 16*half + 8
    const _Float16* ap0 = sT + (size_t)(k0 + row)      * N_OBS + half * 16;
    const _Float16* ap1 = sT + (size_t)(k0 + 16 + row) * N_OBS + half * 16;
    // 16-bit B 32x16: lane holds N=lane%16; halves h: K = 16*half + h
    //   -> one contiguous 16-half run at n0 + 16*half
    const _Float16* bp0 = sT + (size_t)(l0 + row)      * N_OBS + half * 16;
    const _Float16* bp1 = sT + (size_t)(l0 + 16 + row) * N_OBS + half * 16;

    v8f c00 = {}, c01 = {}, c10 = {}, c11 = {};
    for (int n0 = 0; n0 < N_OBS; n0 += 32) {
        union { v16h v; v8h h[2]; } a0, a1;
        a0.h[0] = *(const v8h*)(ap0 + n0);
        a0.h[1] = *(const v8h*)(ap0 + n0 + 8);
        a1.h[0] = *(const v8h*)(ap1 + n0);
        a1.h[1] = *(const v8h*)(ap1 + n0 + 8);
        v16h b0 = *(const v16h*)(bp0 + n0);
        v16h b1 = *(const v16h*)(bp1 + n0);
        c00 = __builtin_amdgcn_wmma_f32_16x16x32_f16(false, a0.v, false, b0,
                                                     (short)0, c00, false, false);
        c01 = __builtin_amdgcn_wmma_f32_16x16x32_f16(false, a0.v, false, b1,
                                                     (short)0, c01, false, false);
        c10 = __builtin_amdgcn_wmma_f32_16x16x32_f16(false, a1.v, false, b0,
                                                     (short)0, c10, false, false);
        c11 = __builtin_amdgcn_wmma_f32_16x16x32_f16(false, a1.v, false, b1,
                                                     (short)0, c11, false, false);
    }

    const float inv_n = 1.0f / (float)N_OBS;
    #pragma unroll
    for (int r = 0; r < 8; ++r) {
        const size_t rk0 = (size_t)(k0 + r + half * 8) * KH;
        const size_t rk1 = (size_t)(k0 + 16 + r + half * 8) * KH;
        M[rk0 + l0 + row]      = c00[r] * inv_n;
        M[rk0 + l0 + 16 + row] = c01[r] * inv_n;
        M[rk1 + l0 + row]      = c10[r] * inv_n;
        M[rk1 + l0 + 16 + row] = c11[r] * inv_n;
    }
}

// ---------------------------------------------------------------------------
// Kernel 3: A[d][k][l] = M[k][l] * w[k][d] * w[l][d]
// block = (k fixed, 256 consecutive l); coalesced 1KB stores per d.
// HBM traffic ~= mandatory 256 MB write; M read once; wT stays in L2.
// ---------------------------------------------------------------------------
__global__ __launch_bounds__(256) void k_expand(const float* __restrict__ M,
                                                const float* __restrict__ w,
                                                const float* __restrict__ wT,
                                                float* __restrict__ out) {
    __shared__ float wk[DIM_IN];
    const int k = blockIdx.y;
    const int l = blockIdx.x * 256 + threadIdx.x;
    if (threadIdx.x < DIM_IN) wk[threadIdx.x] = w[(size_t)k * DIM_IN + threadIdx.x];
    __syncthreads();

    const float m = M[(size_t)k * KH + l];
    #pragma unroll 4
    for (int d = 0; d < DIM_IN; ++d) {
        out[(size_t)d * (KH * KH) + (size_t)k * KH + l] = m * wk[d] * wT[d * KH + l];
    }
}

// ---------------------------------------------------------------------------
extern "C" void kernel_launch(void* const* d_in, const int* in_sizes, int n_in,
                              void* d_out, int out_size, void* d_ws, size_t ws_size,
                              hipStream_t stream) {
    (void)in_sizes; (void)n_in; (void)out_size; (void)ws_size;
    const float* x = (const float*)d_in[0];   // (2048, 64)
    const float* w = (const float*)d_in[1];   // (1024, 64)
    const float* b = (const float*)d_in[2];   // (1024,)
    float* out = (float*)d_out;               // (64, 1024, 1024)

    char* ws = (char*)d_ws;
    _Float16* sT  = (_Float16*)ws;                        // 1024*2048*2 = 4 MB
    float*    Mb  = (float*)(ws + 4u * 1024 * 1024);      // 1024*1024*4 = 4 MB
    float*    wT  = (float*)(ws + 8u * 1024 * 1024);      // 64*1024*4 = 256 KB

    k_wt    <<<256,  256, 0, stream>>>(w, wT);
    k_s     <<<1024, 256, 0, stream>>>(x, w, b, sT);
    k_gram  <<<128,  256, 0, stream>>>(sT, Mb);           // 1024 32x32 tiles
    k_expand<<<dim3(4, 1024), 256, 0, stream>>>(Mb, w, wT, out);
}
